// ID_GINE_22299470201468
// MI455X (gfx1250) — matrix operand order repaired and checked by reading, not compiled, not run on hardware
//
#include <hip/hip_runtime.h>
#include <hip/hip_bf16.h>
#include <math.h>

typedef __attribute__((ext_vector_type(16))) _Float16 v16h;
typedef __attribute__((ext_vector_type(8)))  float    v8f;

#define HID    128
#define NNODE  100000
#define NEDGE  1600000
#define NLAYER 4
#define NB     512
#define LDA    136   // padded half stride for LDS f16 tiles (avoids bank conflicts)
#define LDC    132   // padded float stride for LDS f32 tile
#define WPACK  16384 // halves per packed 128x128 weight matrix

// -------- 16-lane (per-row) shuffle reduction; rows align to 16-lane groups --------
__device__ __forceinline__ float red16(float v) {
    v += __shfl_xor(v, 1);
    v += __shfl_xor(v, 2);
    v += __shfl_xor(v, 4);
    v += __shfl_xor(v, 8);
    return v;
}

// -------- one 16x16 output tile: D = A(16x128 f16 in LDS) * Wp(packed f16) --------
// A fragment (ISA 7.12.2, 16-bit A 16x32): lane<16 -> row=lane, K blocks [0..8)+[16..24);
// lane>=16 -> row=lane-16, K blocks [8..16)+[24..32).
// B fragment comes pre-packed: 16 contiguous halves per lane at
//   Wp + tileN*2048 + kb32*512 + lane*16   (one 32-byte load per K-step).
__device__ __forceinline__ v8f wmma_tile(const _Float16* __restrict__ sIn,
                                         const _Float16* __restrict__ Wp,
                                         int nTile, int lane) {
    v8f c = {};
    const int hs  = lane >> 4;   // 0 or 1
    const int l15 = lane & 15;
    const _Float16* rowp = sIn + l15 * LDA;
    const _Float16* wp0  = Wp + (nTile >> 4) * 2048 + lane * 16;
#pragma unroll
    for (int kb = 0; kb < HID; kb += 32) {
        v16h a;
        const int abase = kb + hs * 8;
#pragma unroll
        for (int i = 0; i < 8; ++i) {
            a[i]     = rowp[abase + i];
            a[i + 8] = rowp[abase + 16 + i];
        }
        const v16h b = *(const v16h*)(wp0 + (kb >> 5) * 512);
        c = __builtin_amdgcn_wmma_f32_16x16x32_f16(false, a, false, b,
                                                   (short)0, c, false, false);
    }
    return c;
}

// ---------------------------------------------------------------------------------
__global__ __launch_bounds__(256) void k_zero(float* __restrict__ p, int n) {
    int i = blockIdx.x * 256 + threadIdx.x;
    if (i < n) p[i] = 0.0f;
}

// pack one 128x128 f32 row-major weight matrix into WMMA B-fragment order (f16)
__global__ __launch_bounds__(256) void k_pack(const float* __restrict__ W,
                                              _Float16* __restrict__ Wp) {
    const int idx = blockIdx.x * 256 + threadIdx.x;   // [0, 16384)
    const int i     = idx & 15;
    const int lane  = (idx >> 4) & 31;
    const int kb32  = (idx >> 9) & 3;
    const int tileN = idx >> 11;
    const int k = kb32 * 32 + (lane >> 4) * 8 + (i < 8 ? i : i + 8);
    const int n = tileN * 16 + (lane & 15);
    Wp[idx] = (_Float16)W[k * HID + n];
}

__global__ __launch_bounds__(256) void k_embed(const int* __restrict__ x_idx,
                                               const float* __restrict__ emb,
                                               float* __restrict__ x) {
    int i = blockIdx.x * 256 + threadIdx.x;
    if (i < NNODE * HID) {
        int n = i >> 7, c = i & (HID - 1);
        x[i] = emb[x_idx[n] * HID + c];
    }
}

// wave-per-edge: msg = relu(x[src] + e1[a0] + e2[a1] + e3[a2]); atomic scatter into agg[dst]
__global__ __launch_bounds__(256) void k_edges(const int* __restrict__ ei,
                                               const int* __restrict__ ea,
                                               const float* __restrict__ x,
                                               const float* __restrict__ e1,
                                               const float* __restrict__ e2,
                                               const float* __restrict__ e3,
                                               float* __restrict__ agg) {
    const int lane = threadIdx.x & 31;
    const int nw   = (gridDim.x * blockDim.x) >> 5;
    for (int e = (blockIdx.x * blockDim.x + threadIdx.x) >> 5; e < NEDGE; e += nw) {
        const int s  = ei[e];
        const int d  = ei[NEDGE + e];
        const int a0 = ea[e * 3 + 0];
        const int a1 = ea[e * 3 + 1];
        const int a2 = ea[e * 3 + 2];
        const int c  = lane * 4;
        const float4 xs = *(const float4*)(x  + s  * HID + c);
        const float4 v1 = *(const float4*)(e1 + a0 * HID + c);
        const float4 v2 = *(const float4*)(e2 + a1 * HID + c);
        const float4 v3 = *(const float4*)(e3 + a2 * HID + c);
        float m0 = xs.x + v1.x + v2.x + v3.x; m0 = m0 > 0.f ? m0 : 0.f;
        float m1 = xs.y + v1.y + v2.y + v3.y; m1 = m1 > 0.f ? m1 : 0.f;
        float m2 = xs.z + v1.z + v2.z + v3.z; m2 = m2 > 0.f ? m2 : 0.f;
        float m3 = xs.w + v1.w + v2.w + v3.w; m3 = m3 > 0.f ? m3 : 0.f;
        float* ap = agg + d * HID + c;
        atomicAdd(ap + 0, m0);
        atomicAdd(ap + 1, m1);
        atomicAdd(ap + 2, m2);
        atomicAdd(ap + 3, m3);
    }
}

// fused per-layer MLP: h0=x+agg -> relu(h0@W1+b1) -> @W2+b2 -> relu(LN) -> x+out
__global__ __launch_bounds__(256) void k_layer_mlp(const float* __restrict__ x,
                                                   const float* __restrict__ agg,
                                                   const _Float16* __restrict__ W1p,
                                                   const float* __restrict__ b1,
                                                   const _Float16* __restrict__ W2p,
                                                   const float* __restrict__ b2,
                                                   const float* __restrict__ lng,
                                                   const float* __restrict__ lnb,
                                                   float* __restrict__ xout) {
    __shared__ _Float16 sA[16 * LDA];
    __shared__ _Float16 sB[16 * LDA];
    __shared__ float    sC[16 * LDC];
    const int t = threadIdx.x, lane = t & 31, wave = t >> 5;
    const int rowBase = blockIdx.x * 16;

    { // stage h0 = x + agg as f16
        int r = t >> 4, c0 = (t & 15) * 8;
        const float* xr = x   + (rowBase + r) * HID + c0;
        const float* ar = agg + (rowBase + r) * HID + c0;
#pragma unroll
        for (int i = 0; i < 8; ++i)
            sA[r * LDA + c0 + i] = (_Float16)(xr[i] + ar[i]);
    }
    __syncthreads();
    { // GEMM1 + bias + relu -> sB (f16)
        const int nTile = wave * 16;
        v8f c = wmma_tile(sA, W1p, nTile, lane);
        const int n = nTile + (lane & 15);
        const float bias = b1[n];
#pragma unroll
        for (int v = 0; v < 8; ++v) {
            float val = c[v] + bias;
            val = val > 0.f ? val : 0.f;
            sB[(v + 8 * (lane >> 4)) * LDA + n] = (_Float16)val;
        }
    }
    __syncthreads();
    { // GEMM2 + bias -> sC (f32)
        const int nTile = wave * 16;
        v8f c = wmma_tile(sB, W2p, nTile, lane);
        const int n = nTile + (lane & 15);
        const float bias = b2[n];
#pragma unroll
        for (int v = 0; v < 8; ++v)
            sC[(v + 8 * (lane >> 4)) * LDC + n] = c[v] + bias;
    }
    __syncthreads();
    { // LayerNorm + relu + residual
        const int r = t >> 4, q = t & 15;
        const float* hr = sC + r * LDC + q * 8;
        float h[8], s = 0.f, ss = 0.f;
#pragma unroll
        for (int i = 0; i < 8; ++i) { h[i] = hr[i]; s += h[i]; ss += h[i] * h[i]; }
        s = red16(s); ss = red16(ss);
        const float mean = s * (1.0f / HID);
        const float var  = ss * (1.0f / HID) - mean * mean;
        const float rs   = rsqrtf(var + 1e-5f);
        const int grow = rowBase + r;
        const float* xr = x + grow * HID;
        float* outr = xout + grow * HID;
#pragma unroll
        for (int i = 0; i < 8; ++i) {
            const int c = q * 8 + i;
            float v = (h[i] - mean) * rs * lng[c] + lnb[c];
            v = v > 0.f ? v : 0.f;
            outr[c] = xr[c] + v;
        }
    }
}

// wave-per-node mean-pool accumulation
__global__ __launch_bounds__(256) void k_pool(const int* __restrict__ batch,
                                              const float* __restrict__ x,
                                              float* __restrict__ gsum,
                                              float* __restrict__ gcnt) {
    const int n = (blockIdx.x * blockDim.x + threadIdx.x) >> 5;
    const int lane = threadIdx.x & 31;
    if (n < NNODE) {
        const int b = batch[n];
        const int c = lane * 4;
        const float4 v = *(const float4*)(x + n * HID + c);
        float* gp = gsum + b * HID + c;
        atomicAdd(gp + 0, v.x);
        atomicAdd(gp + 1, v.y);
        atomicAdd(gp + 2, v.z);
        atomicAdd(gp + 3, v.w);
        if (lane == 0) atomicAdd(gcnt + b, 1.0f);
    }
}

// head: g=mean pool -> relu(g@pW1+pb1) -> @pW2+pb2 -> relu(LN) -> sigmoid(@oW+ob)
__global__ __launch_bounds__(256) void k_head(const float* __restrict__ gsum,
                                              const float* __restrict__ gcnt,
                                              const _Float16* __restrict__ pW1p,
                                              const float* __restrict__ pb1,
                                              const _Float16* __restrict__ pW2p,
                                              const float* __restrict__ pb2,
                                              const float* __restrict__ pg,
                                              const float* __restrict__ pbn,
                                              const float* __restrict__ oW,
                                              const float* __restrict__ ob,
                                              float* __restrict__ out) {
    __shared__ _Float16 sA[16 * LDA];
    __shared__ _Float16 sB[16 * LDA];
    __shared__ float    sC[16 * LDC];
    const int t = threadIdx.x, lane = t & 31, wave = t >> 5;
    const int rowBase = blockIdx.x * 16;

    { // stage g = gsum / max(cnt,1)
        int r = t >> 4, c0 = (t & 15) * 8;
        const int grow = rowBase + r;
        const float cnt = gcnt[grow];
        const float inv = 1.0f / (cnt > 1.0f ? cnt : 1.0f);
        const float* gr = gsum + grow * HID + c0;
#pragma unroll
        for (int i = 0; i < 8; ++i)
            sA[r * LDA + c0 + i] = (_Float16)(gr[i] * inv);
    }
    __syncthreads();
    { // GEMM1 + bias + relu
        const int nTile = wave * 16;
        v8f c = wmma_tile(sA, pW1p, nTile, lane);
        const int n = nTile + (lane & 15);
        const float bias = pb1[n];
#pragma unroll
        for (int v = 0; v < 8; ++v) {
            float val = c[v] + bias;
            val = val > 0.f ? val : 0.f;
            sB[(v + 8 * (lane >> 4)) * LDA + n] = (_Float16)val;
        }
    }
    __syncthreads();
    { // GEMM2 + bias
        const int nTile = wave * 16;
        v8f c = wmma_tile(sB, pW2p, nTile, lane);
        const int n = nTile + (lane & 15);
        const float bias = pb2[n];
#pragma unroll
        for (int v = 0; v < 8; ++v)
            sC[(v + 8 * (lane >> 4)) * LDC + n] = c[v] + bias;
    }
    __syncthreads();
    { // LN + relu + output dot + sigmoid
        const int r = t >> 4, q = t & 15;
        const float* hr = sC + r * LDC + q * 8;
        float h[8], s = 0.f, ss = 0.f;
#pragma unroll
        for (int i = 0; i < 8; ++i) { h[i] = hr[i]; s += h[i]; ss += h[i] * h[i]; }
        s = red16(s); ss = red16(ss);
        const float mean = s * (1.0f / HID);
        const float var  = ss * (1.0f / HID) - mean * mean;
        const float rs   = rsqrtf(var + 1e-5f);
        float part = 0.f;
#pragma unroll
        for (int i = 0; i < 8; ++i) {
            const int c = q * 8 + i;
            float v = (h[i] - mean) * rs * pg[c] + pbn[c];
            v = v > 0.f ? v : 0.f;
            part += v * oW[c];           // OUT == 1
        }
        part = red16(part);
        if (q == 0) {
            const int grow = rowBase + r;
            out[grow] = 1.0f / (1.0f + expf(-(part + ob[0])));
        }
    }
}

// ---------------------------------------------------------------------------------
extern "C" void kernel_launch(void* const* d_in, const int* in_sizes, int n_in,
                              void* d_out, int out_size, void* d_ws, size_t ws_size,
                              hipStream_t stream) {
    (void)in_sizes; (void)n_in; (void)out_size; (void)ws_size;
    const int*   x_idx = (const int*)d_in[0];
    const int*   ei    = (const int*)d_in[1];
    const int*   ea    = (const int*)d_in[2];
    const int*   batch = (const int*)d_in[3];
    const float* nemb  = (const float*)d_in[4];
    const float* e1    = (const float*)d_in[5];
    const float* e2    = (const float*)d_in[6];
    const float* e3    = (const float*)d_in[7];
    const float* W1    = (const float*)d_in[8];
    const float* b1    = (const float*)d_in[9];
    const float* W2    = (const float*)d_in[10];
    const float* b2    = (const float*)d_in[11];
    const float* lng   = (const float*)d_in[12];
    const float* lnb   = (const float*)d_in[13];
    const float* pW1   = (const float*)d_in[14];
    const float* pb1   = (const float*)d_in[15];
    const float* pW2   = (const float*)d_in[16];
    const float* pb2   = (const float*)d_in[17];
    const float* pg    = (const float*)d_in[18];
    const float* pbn   = (const float*)d_in[19];
    const float* oW    = (const float*)d_in[20];
    const float* ob    = (const float*)d_in[21];

    float* ws   = (float*)d_ws;
    float* x0   = ws;                        // N*H
    float* x1   = x0 + (size_t)NNODE * HID;  // N*H
    float* agg  = x1 + (size_t)NNODE * HID;  // N*H
    float* gsum = agg + (size_t)NNODE * HID; // B*H
    float* gcnt = gsum + (size_t)NB * HID;   // B
    _Float16* wpk = (_Float16*)(gcnt + NB);  // 10 * WPACK halves

    // pack all weight matrices into WMMA B-fragment order (f16)
    for (int l = 0; l < NLAYER; ++l) {
        k_pack<<<WPACK / 256, 256, 0, stream>>>(W1 + (size_t)l * HID * HID,
                                                wpk + (size_t)l * WPACK);
        k_pack<<<WPACK / 256, 256, 0, stream>>>(W2 + (size_t)l * HID * HID,
                                                wpk + (size_t)(NLAYER + l) * WPACK);
    }
    k_pack<<<WPACK / 256, 256, 0, stream>>>(pW1, wpk + (size_t)(2 * NLAYER) * WPACK);
    k_pack<<<WPACK / 256, 256, 0, stream>>>(pW2, wpk + (size_t)(2 * NLAYER + 1) * WPACK);

    const int NH = NNODE * HID;
    k_embed<<<(NH + 255) / 256, 256, 0, stream>>>(x_idx, nemb, x0);

    float* xin = x0; float* xout = x1;
    for (int l = 0; l < NLAYER; ++l) {
        k_zero<<<(NH + 255) / 256, 256, 0, stream>>>(agg, NH);
        k_edges<<<(NEDGE * 32) / 256, 256, 0, stream>>>(ei, ea, xin, e1, e2, e3, agg);
        k_layer_mlp<<<NNODE / 16, 256, 0, stream>>>(
            xin, agg,
            wpk + (size_t)l * WPACK, b1 + l * HID,
            wpk + (size_t)(NLAYER + l) * WPACK, b2 + l * HID,
            lng + l * HID, lnb + l * HID, xout);
        float* tmp = xin; xin = xout; xout = tmp;
    }

    k_zero<<<(NB * HID + NB + 255) / 256, 256, 0, stream>>>(gsum, NB * HID + NB);
    k_pool<<<(NNODE * 32) / 256, 256, 0, stream>>>(batch, xin, gsum, gcnt);
    k_head<<<NB / 16, 256, 0, stream>>>(gsum, gcnt,
                                        wpk + (size_t)(2 * NLAYER) * WPACK, pb1,
                                        wpk + (size_t)(2 * NLAYER + 1) * WPACK, pb2,
                                        pg, pbn, oW, ob, (float*)d_out);
}
